// FastSelfAttention_38285338477061
// MI455X (gfx1250) — matrix operand order repaired and checked by reading, not compiled, not run on hardware
//
#include <hip/hip_runtime.h>
#include <hip/hip_bf16.h>
#include <math.h>

#define DIM   1024
#define S_LEN 2048
#define HEADS 16
#define DH    64
#define BM    128
#define BN    128
#define BK    32
#define NK    (DIM / BK)

typedef __bf16 bf16;
typedef bf16  bf16x4  __attribute__((ext_vector_type(4)));
typedef bf16  bf16x8  __attribute__((ext_vector_type(8)));
typedef bf16  bf16x16 __attribute__((ext_vector_type(16)));
typedef float v8f     __attribute__((ext_vector_type(8)));

// Async global->LDS 16-byte copy (CDNA5, tracked with ASYNCcnt).
__device__ __forceinline__ void async_copy_b128(unsigned lds_off, const void* gaddr) {
  asm volatile("global_load_async_to_lds_b128 %0, %1, off"
               :: "v"(lds_off), "v"(gaddr) : "memory");
}
__device__ __forceinline__ void wait_asynccnt0() {
  asm volatile("s_wait_asynccnt 0x0" ::: "memory");
}

// ---------------------------------------------------------------------------
// One-shot: W [1024x1024] f32 row-major [k][n]  ->  Wt bf16 [n][k]
// ---------------------------------------------------------------------------
__global__ __launch_bounds__(256)
void transpose_w_bf16(const float* __restrict__ W, bf16* __restrict__ Wt) {
  __shared__ float t[32][33];
  const int tid = threadIdx.x;
  const int n0 = blockIdx.x * 32, k0 = blockIdx.y * 32;
  {
    int r = tid >> 3, c = (tid & 7) * 4;
    float4 v = *(const float4*)(W + (size_t)(k0 + r) * DIM + n0 + c);
    t[r][c + 0] = v.x; t[r][c + 1] = v.y; t[r][c + 2] = v.z; t[r][c + 3] = v.w;
  }
  __syncthreads();
  {
    int n = tid >> 3, k = (tid & 7) * 4;
    bf16x4 o = { (bf16)t[k + 0][n], (bf16)t[k + 1][n],
                 (bf16)t[k + 2][n], (bf16)t[k + 3][n] };
    *(bf16x4*)(Wt + (size_t)(n0 + n) * DIM + k0 + k) = o;
  }
}

// ---------------------------------------------------------------------------
// Elementwise f32 -> bf16, optional per-(b,d) scale
// ---------------------------------------------------------------------------
template<bool SCALED>
__global__ __launch_bounds__(256)
void convert_bf16(const float* __restrict__ in, const float* __restrict__ sc,
                  bf16* __restrict__ out, size_t total4) {
  size_t i = (size_t)blockIdx.x * blockDim.x + threadIdx.x;
  if (i >= total4) return;
  size_t f = i * 4;
  float4 v = *(const float4*)(in + f);
  if (SCALED) {
    int d = (int)(f & (DIM - 1));
    int b = (int)(f >> 21);                              // / (S_LEN*DIM)
    const float* s = sc + (size_t)b * DIM + d;
    v.x *= s[0]; v.y *= s[1]; v.z *= s[2]; v.w *= s[3];
  }
  bf16x4 o = { (bf16)v.x, (bf16)v.y, (bf16)v.z, (bf16)v.w };
  *(bf16x4*)(out + f) = o;
}

// ---------------------------------------------------------------------------
// Big GEMM: out[m,n] = sum_k A[m,k]*Wt[n,k] + bias[n] (+ addsrc[m,n])
// A: [M x 1024] bf16 [m][k];  Wt: [1024 x 1024] bf16 [n][k].
// Block 256 thr (8 waves), tile 128x128, K-step 32; wave = 32x64 (2x4 WMMA).
// Double-buffered LDS, explicit 2x software pipeline, async global->LDS staging.
// ---------------------------------------------------------------------------
template<bool HAS_ADD>
__global__ __launch_bounds__(256)
void gemm_bf16_wmma(const bf16* __restrict__ A, const bf16* __restrict__ Wt,
                    const float* __restrict__ bias,
                    const float* __restrict__ addsrc,
                    float* __restrict__ out) {
  __shared__ alignas(32) bf16 As[2][BM * BK];   // [m][k]  2 x 8 KB
  __shared__ alignas(32) bf16 Bs[2][BN * BK];   // [n][k]  2 x 8 KB

  const int tid  = threadIdx.x;
  const int m0   = blockIdx.x * BM;
  const int n0   = blockIdx.y * BN;
  const int wave = tid >> 5;
  const int lane = tid & 31;
  const int wr   = wave & 3;
  const int wc   = wave >> 2;
  const int l16  = lane & 15;
  const int lhalf = lane >> 4;

  // per-thread staging coordinates: two 8-elem chunks per matrix
  const int row0 = tid >> 2;                   // 0..63
  const int row1 = row0 + 64;                  // 64..127
  const int cc   = (tid & 3) * 8;              // 0,8,16,24
  const bf16* pA0 = A  + (size_t)(m0 + row0) * DIM + cc;
  const bf16* pA1 = A  + (size_t)(m0 + row1) * DIM + cc;
  const bf16* pB0 = Wt + (size_t)(n0 + row0) * DIM + cc;
  const bf16* pB1 = Wt + (size_t)(n0 + row1) * DIM + cc;
  const unsigned lA0 = (unsigned)(uintptr_t)&As[0][row0 * BK + cc];
  const unsigned lA1 = (unsigned)(uintptr_t)&As[0][row1 * BK + cc];
  const unsigned lB0 = (unsigned)(uintptr_t)&Bs[0][row0 * BK + cc];
  const unsigned lB1 = (unsigned)(uintptr_t)&Bs[0][row1 * BK + cc];
  const unsigned BUFSTEP = (unsigned)(BM * BK * sizeof(bf16));   // 8192 B

  v8f acc[2][4];
#pragma unroll
  for (int mi = 0; mi < 2; ++mi)
#pragma unroll
    for (int ni = 0; ni < 4; ++ni) acc[mi][ni] = (v8f)0.0f;

  // compile-time buffer index in both macros -> no parity phis, no reg copies
#define STAGE(BUF, K0)                                              \
  do {                                                              \
    async_copy_b128(lA0 + (BUF) * BUFSTEP, pA0 + (K0));             \
    async_copy_b128(lA1 + (BUF) * BUFSTEP, pA1 + (K0));             \
    async_copy_b128(lB0 + (BUF) * BUFSTEP, pB0 + (K0));             \
    async_copy_b128(lB1 + (BUF) * BUFSTEP, pB1 + (K0));             \
  } while (0)

#define COMPUTE(BUF)                                                             \
  do {                                                                           \
    bf16x16 afrag[2], bfrag[4];                                                  \
    _Pragma("unroll")                                                            \
    for (int mi = 0; mi < 2; ++mi) {                                             \
      int row = wr * 32 + mi * 16 + l16;                                         \
      afrag[mi] = *(const bf16x16*)&As[BUF][row * BK + lhalf * 16];              \
    }                                                                            \
    _Pragma("unroll")                                                            \
    for (int ni = 0; ni < 4; ++ni) {                                             \
      int col = wc * 64 + ni * 16 + l16;                                         \
      bfrag[ni] = *(const bf16x16*)&Bs[BUF][col * BK + lhalf * 16];              \
    }                                                                            \
    _Pragma("unroll")                                                            \
    for (int mi = 0; mi < 2; ++mi)                                               \
      _Pragma("unroll")                                                          \
      for (int ni = 0; ni < 4; ++ni)                                             \
        acc[mi][ni] = __builtin_amdgcn_wmma_f32_16x16x32_bf16(                   \
            false, afrag[mi], false, bfrag[ni], (short)0, acc[mi][ni],           \
            false, false);                                                       \
  } while (0)

  // prologue: tile 0 -> buf0
  STAGE(0, 0);
  wait_asynccnt0();
  __syncthreads();

  // steady state: tiles it (buf0) / it+1 (buf1); stage it+1 (buf1) / it+2 (buf0)
#pragma unroll 1
  for (int it = 0; it < NK - 2; it += 2) {
    STAGE(1, (it + 1) * BK);
    COMPUTE(0);
    wait_asynccnt0();
    __syncthreads();
    STAGE(0, (it + 2) * BK);
    COMPUTE(1);
    wait_asynccnt0();
    __syncthreads();
  }
  // tail: buf0 holds tile NK-2
  STAGE(1, (NK - 1) * BK);
  COMPUTE(0);
  wait_asynccnt0();
  __syncthreads();
  COMPUTE(1);
#undef STAGE
#undef COMPUTE

  // ---- epilogue: VGPR j -> lanes0-15 M=j, lanes16-31 M=8+j ----
#pragma unroll
  for (int mi = 0; mi < 2; ++mi)
#pragma unroll
    for (int ni = 0; ni < 4; ++ni) {
      int gm = m0 + wr * 32 + mi * 16 + 8 * lhalf;
      int gn = n0 + wc * 64 + ni * 16 + l16;
      float bn = bias[gn];
#pragma unroll
      for (int j = 0; j < 8; ++j) {
        size_t off = (size_t)(gm + j) * DIM + gn;
        float v = acc[mi][ni][j] + bn;
        if (HAS_ADD) v += addsrc[off];
        out[off] = v;
      }
    }
}

// ---------------------------------------------------------------------------
// scores[b,h,s] = (Min[m,:]·Wa[:,h] + ba[h]) / 8 + mask[b,s],  m = b*S+s
// ---------------------------------------------------------------------------
__global__ __launch_bounds__(256)
void scores_kernel(const float* __restrict__ Min, const float* __restrict__ Wa,
                   const float* __restrict__ ba, const float* __restrict__ mask,
                   float* __restrict__ scores) {
  __shared__ float At[16 * DIM];               // 64 KB
  const int tid = threadIdx.x;
  const size_t base = (size_t)blockIdx.x * 16 * DIM;
#pragma unroll
  for (int i = 0; i < 16; ++i) {
    int f = (tid + i * 256) * 4;
    *(float4*)&At[f] = *(const float4*)(Min + base + f);
  }
  __syncthreads();

  const int r = tid >> 4, h = tid & 15;
  const float* arow = &At[r * DIM];
  float acc = 0.f;
#pragma unroll 4
  for (int k = 0; k < DIM; ++k)
    acc = fmaf(arow[k], Wa[k * HEADS + h], acc);

  const int m = blockIdx.x * 16 + r;
  const int b = m / S_LEN, sl = m % S_LEN;
  scores[((size_t)b * HEADS + h) * S_LEN + sl] =
      (acc + ba[h]) * 0.125f + mask[(size_t)b * S_LEN + sl];
}

// ---------------------------------------------------------------------------
// One block per (b,h): softmax over S, then pooled[b,h,dh]=sum_s w_s*Min[b,s,h*64+dh]
// ---------------------------------------------------------------------------
__global__ __launch_bounds__(256)
void softmax_pool(const float* __restrict__ scores, const float* __restrict__ Min,
                  float* __restrict__ pooled) {
  __shared__ float w[S_LEN];
  __shared__ float red[256];
  __shared__ float inv_sum;
  const int tid = threadIdx.x;
  const int b = blockIdx.x >> 4, h = blockIdx.x & 15;
  const float* sc = scores + ((size_t)b * HEADS + h) * S_LEN;

  float mx = -1e30f;
  for (int i = tid; i < S_LEN; i += 256) { float v = sc[i]; w[i] = v; mx = fmaxf(mx, v); }
  red[tid] = mx; __syncthreads();
  for (int st = 128; st > 0; st >>= 1) {
    if (tid < st) red[tid] = fmaxf(red[tid], red[tid + st]);
    __syncthreads();
  }
  mx = red[0]; __syncthreads();

  float sum = 0.f;
  for (int i = tid; i < S_LEN; i += 256) { float e = __expf(w[i] - mx); w[i] = e; sum += e; }
  red[tid] = sum; __syncthreads();
  for (int st = 128; st > 0; st >>= 1) {
    if (tid < st) red[tid] += red[tid + st];
    __syncthreads();
  }
  if (tid == 0) inv_sum = 1.f / red[0];
  __syncthreads();

  const int dh = tid & 63, g = tid >> 6;
  float acc = 0.f;
  const float* mb = Min + (size_t)b * S_LEN * DIM + h * DH + dh;
  for (int s = g; s < S_LEN; s += 4)
    acc = fmaf(w[s], mb[(size_t)s * DIM], acc);
  red[tid] = acc; __syncthreads();
  if (g == 0) {
    float tot = red[tid] + red[tid + 64] + red[tid + 128] + red[tid + 192];
    pooled[(size_t)b * DIM + h * DH + dh] = tot * inv_sum;
  }
}

// ---------------------------------------------------------------------------
// In-place: M[b,s,d] *= pq[b,d]
// ---------------------------------------------------------------------------
__global__ __launch_bounds__(256)
void modulate(float* __restrict__ M, const float* __restrict__ pq, size_t total4) {
  size_t i = (size_t)blockIdx.x * blockDim.x + threadIdx.x;
  if (i >= total4) return;
  size_t f = i * 4;
  int d = (int)(f & (DIM - 1));
  int b = (int)(f >> 21);
  const float* s = pq + (size_t)b * DIM + d;
  float4 v = *(float4*)(M + f);
  v.x *= s[0]; v.y *= s[1]; v.z *= s[2]; v.w *= s[3];
  *(float4*)(M + f) = v;
}

extern "C" void kernel_launch(void* const* d_in, const int* in_sizes, int n_in,
                              void* d_out, int out_size, void* d_ws, size_t ws_size,
                              hipStream_t stream) {
  const float* hs   = (const float*)d_in[0];
  const float* mask = (const float*)d_in[1];
  const float* Wq   = (const float*)d_in[2];
  const float* bq   = (const float*)d_in[3];
  const float* Wqa  = (const float*)d_in[4];
  const float* bqa  = (const float*)d_in[5];
  const float* Wk   = (const float*)d_in[6];
  const float* bk   = (const float*)d_in[7];
  const float* Wka  = (const float*)d_in[8];
  const float* bka  = (const float*)d_in[9];
  const float* Wt   = (const float*)d_in[10];
  const float* bt   = (const float*)d_in[11];
  float* out = (float*)d_out;

  const int M = in_sizes[0] / DIM;             // B*S = 32768
  const int B = M / S_LEN;
  const size_t md = (size_t)M * DIM;

  char* p = (char*)d_ws;
  float* mixed_q  = (float*)p;            p += md * 4;
  float* mixed_k  = (float*)p;            p += md * 4;
  float* scores   = (float*)p;            p += (size_t)B * HEADS * S_LEN * 4;
  float* pooled_q = (float*)p;            p += (size_t)B * DIM * 4;
  float* pooled_k = (float*)p;            p += (size_t)B * DIM * 4;
  bf16*  WqT      = (bf16*)p;             p += (size_t)DIM * DIM * 2;
  bf16*  WkT      = (bf16*)p;             p += (size_t)DIM * DIM * 2;
  bf16*  WtT      = (bf16*)p;             p += (size_t)DIM * DIM * 2;
  bf16*  abf      = (bf16*)p;             p += md * 2;   // hs_bf, later weighted_bf

  dim3 gg(M / BM, DIM / BN);                   // (256, 8)
  dim3 gt(DIM / 32, DIM / 32);
  const int score_blocks = M / 16;
  const int bh_blocks = B * HEADS;
  const size_t total4 = md / 4;
  const int cb = (int)((total4 + 255) / 256);

  transpose_w_bf16<<<gt, 256, 0, stream>>>(Wq, WqT);
  transpose_w_bf16<<<gt, 256, 0, stream>>>(Wk, WkT);
  transpose_w_bf16<<<gt, 256, 0, stream>>>(Wt, WtT);
  convert_bf16<false><<<cb, 256, 0, stream>>>(hs, nullptr, abf, total4);
  gemm_bf16_wmma<false><<<gg, 256, 0, stream>>>(abf, WqT, bq, nullptr, mixed_q);
  gemm_bf16_wmma<false><<<gg, 256, 0, stream>>>(abf, WkT, bk, nullptr, mixed_k);
  scores_kernel<<<score_blocks, 256, 0, stream>>>(mixed_q, Wqa, bqa, mask, scores);
  softmax_pool<<<bh_blocks, 256, 0, stream>>>(scores, mixed_q, pooled_q);
  modulate<<<cb, 256, 0, stream>>>(mixed_k, pooled_q, total4);
  scores_kernel<<<score_blocks, 256, 0, stream>>>(mixed_k, Wka, bka, mask, scores);
  softmax_pool<<<bh_blocks, 256, 0, stream>>>(scores, mixed_k, pooled_k);
  convert_bf16<true><<<cb, 256, 0, stream>>>(mixed_q, pooled_k, abf, total4);
  gemm_bf16_wmma<true><<<gg, 256, 0, stream>>>(abf, WtT, bt, mixed_q, out);
}
